// MaskedMultiHeadAttention_74131135529739
// MI455X (gfx1250) — matrix operand order repaired and checked by reading, not compiled, not run on hardware
//
#include <hip/hip_runtime.h>
#include <hip/hip_bf16.h>

#define SEQ     2048
#define DMODEL  1024
#define HEADS   16
#define DK      64
#define BATCH   2
#define MTOT    (BATCH*SEQ)     // 4096
#define NQKV    (3*DMODEL)      // 3072

typedef __attribute__((ext_vector_type(16))) __bf16 v16bf;
typedef __attribute__((ext_vector_type(8)))  float  v8f;
typedef __attribute__((ext_vector_type(4)))  int    v4i;

union Frag16 {
  v16bf v;
  unsigned short h[16];
  uint4 q[2];
};

// ---- gfx1250 async global->LDS support (guarded, with safe fallback) ------
#if defined(__has_builtin)
#  if __has_builtin(__builtin_amdgcn_global_load_async_to_lds_b128)
#    define HAVE_ASYNC_LDS 1
#  endif
#  if __has_builtin(__builtin_amdgcn_s_wait_asynccnt)
#    define WAIT_ASYNC(n) __builtin_amdgcn_s_wait_asynccnt(n)
#  endif
#endif
#ifndef HAVE_ASYNC_LDS
#  define HAVE_ASYNC_LDS 0
#endif
#ifndef WAIT_ASYNC
#  if HAVE_ASYNC_LDS
#    define WAIT_ASYNC(n) asm volatile("s_wait_asynccnt %0" :: "i"(n) : "memory")
#  else
#    define WAIT_ASYNC(n) ((void)0)
#  endif
#endif

__device__ __forceinline__ void g2l16(unsigned short* l, const unsigned short* g) {
#if HAVE_ASYNC_LDS
  __builtin_amdgcn_global_load_async_to_lds_b128(
      (__attribute__((address_space(1))) v4i*)(g),
      (__attribute__((address_space(3))) v4i*)(l), 0, 0);
#else
  *(uint4*)l = *(const uint4*)g;
#endif
}

__device__ __forceinline__ unsigned short f2bf(float f) {
  unsigned int u = __float_as_uint(f);
  u += 0x7FFFu + ((u >> 16) & 1u);          // round-to-nearest-even
  return (unsigned short)(u >> 16);
}

// ---------------------------------------------------------------------------
// Kernel 0a: row-major f32 -> bf16 convert (x -> xb), vectorized 4-wide.
// ---------------------------------------------------------------------------
__global__ void convert_bf16(const float* __restrict__ src,
                             unsigned short* __restrict__ dst, int n4) {
  int idx = blockIdx.x * 256 + threadIdx.x;
  if (idx >= n4) return;
  float4 c = ((const float4*)src)[idx];
  ushort4 o;
  o.x = f2bf(c.x); o.y = f2bf(c.y); o.z = f2bf(c.z); o.w = f2bf(c.w);
  ((ushort4*)dst)[idx] = o;
}

// ---------------------------------------------------------------------------
// Kernel 0b: transpose f32 [rows, cols] -> bf16 [cols, rows]
// ---------------------------------------------------------------------------
__global__ void transpose_to_bf16(const float* __restrict__ src,
                                  unsigned short* __restrict__ dst,
                                  int rows, int cols) {
  int idx = blockIdx.x * 256 + threadIdx.x;
  if (idx >= rows * cols) return;
  int r = idx / cols, c = idx % cols;        // coalesced read along cols
  dst[(size_t)c * rows + r] = f2bf(src[idx]);
}

// ---------------------------------------------------------------------------
// Kernel 1: QKV GEMM (bf16 WMMA, f32 accum) + bias + RoPE epilogue.
// Block: 256 thr = 8 waves (2Mx4N), wave tile 64x64 (4x4 frags),
// block tile 128x256.  16 WMMA : 16 b128 loads per k-step.
// ---------------------------------------------------------------------------
__global__ __launch_bounds__(256)
void qkv_gemm_rope(const unsigned short* __restrict__ xb,
                   const unsigned short* __restrict__ wT,
                   const float* __restrict__ bias,
                   unsigned short* __restrict__ qh,
                   unsigned short* __restrict__ kh,
                   unsigned short* __restrict__ vT) {
  const int lane  = threadIdx.x & 31;
  const int wid   = threadIdx.x >> 5;
  const int waveM = wid >> 2, waveN = wid & 3;
  const int l15   = lane & 15;
  const int hi    = lane >> 4;
  const int rowA0 = blockIdx.y * 128 + waveM * 64;
  const int colB0 = blockIdx.x * 256 + waveN * 64;

  v8f acc[4][4];
#pragma unroll
  for (int mi = 0; mi < 4; ++mi)
#pragma unroll
    for (int ni = 0; ni < 4; ++ni) acc[mi][ni] = {};

  for (int kb = 0; kb < DMODEL; kb += 32) {
    Frag16 a[4];
#pragma unroll
    for (int mi = 0; mi < 4; ++mi) {
      const unsigned short* ap = xb + (size_t)(rowA0 + mi * 16 + l15) * DMODEL
                                    + kb + hi * 8;
      a[mi].q[0] = *(const uint4*)ap;         // K = k0..k0+7
      a[mi].q[1] = *(const uint4*)(ap + 16);  // K = k0+16..k0+23
      if (kb + 32 < DMODEL) __builtin_prefetch(ap + 32, 0, 3);
    }
    Frag16 b[4];
#pragma unroll
    for (int ni = 0; ni < 4; ++ni) {
      const unsigned short* bp = wT + (size_t)(colB0 + ni * 16 + l15) * DMODEL
                                    + kb + hi * 8;
      b[ni].q[0] = *(const uint4*)bp;
      b[ni].q[1] = *(const uint4*)(bp + 16);
      if (kb + 32 < DMODEL) __builtin_prefetch(bp + 32, 0, 3);
    }
#pragma unroll
    for (int mi = 0; mi < 4; ++mi)
#pragma unroll
      for (int ni = 0; ni < 4; ++ni)
        acc[mi][ni] = __builtin_amdgcn_wmma_f32_16x16x32_bf16(
            false, a[mi].v, false, b[ni].v, (short)0, acc[mi][ni], false, false);
  }

  // Epilogue: bias, RoPE (q,k), scale q by 8, scatter to head layouts.
#pragma unroll
  for (int mi = 0; mi < 4; ++mi) {
#pragma unroll
    for (int ni = 0; ni < 4; ++ni) {
      const int n  = colB0 + ni * 16 + l15;
      const float bv = bias[n];
      const int region = n >> 10;           // 0=q 1=k 2=v (uniform per fragment)
      const int nn = n & 1023;
      const int h  = nn >> 6, d = nn & 63;
      const int pairi = d >> 1;
      const float invf = __expf(-(float)(2 * pairi) * (9.210340372f / 64.0f));
      const int parity = d & 1;
#pragma unroll
      for (int e = 0; e < 8; ++e) {
        const int m = rowA0 + mi * 16 + e + hi * 8;
        const int bnum = m >> 11, s = m & (SEQ - 1);
        float val = acc[mi][ni][e] + bv;
        if (region < 2) {
          float sn, cs;
          __sincosf((float)s * invf, &sn, &cs);
          const float pv = __shfl_xor(val, 1, 32);  // pair partner = adjacent lane
          float r = (parity == 0) ? (val * cs - pv * sn)
                                  : (pv * sn + val * cs);
          const size_t dst = (((size_t)bnum * HEADS + h) * SEQ + s) * DK + d;
          if (region == 0) qh[dst] = f2bf(r * 8.0f);   // fold sqrt(dk) into Q
          else             kh[dst] = f2bf(r);
        } else {
          vT[(((size_t)bnum * HEADS + h) * DK + d) * SEQ + s] = f2bf(val);
        }
      }
    }
  }
}

// ---------------------------------------------------------------------------
// Kernel 2: causal flash attention.
// 4 waves/block over 4 consecutive q-tiles; K/V 32-key blocks are staged into
// LDS once per block via async global->LDS DMA (double-buffered, ASYNCcnt),
// then all waves read WMMA fragments from LDS (4x less L2 traffic).
// Scores computed transposed (S^T = K x Q^T) so softmax is in-lane and the
// S^T C-fragment repacks into the P A-fragment with zero cross-lane traffic.
// ---------------------------------------------------------------------------
__global__ __launch_bounds__(128)
void flash_attn(const unsigned short* __restrict__ qh,
                const unsigned short* __restrict__ kh,
                const unsigned short* __restrict__ vT,
                unsigned short* __restrict__ concat) {
  // padded LDS tiles: K row stride 72 shorts (144B), V row stride 40 (80B)
  __shared__ __attribute__((aligned(16))) unsigned short Kb[2][32 * 72];
  __shared__ __attribute__((aligned(16))) unsigned short Vb[2][64 * 40];

  const int tid  = threadIdx.x;
  const int lane = tid & 31;
  const int wid  = tid >> 5;
  const int l15  = lane & 15;
  const int hi   = lane >> 4;
  const int bh   = blockIdx.y;
  const int b    = bh >> 4, h = bh & 15;
  const int q_base = (blockIdx.x * 4 + wid) * 16;

  const unsigned short* Q = qh + (size_t)bh * SEQ * DK;
  const unsigned short* K = kh + (size_t)bh * SEQ * DK;
  const unsigned short* V = vT + (size_t)bh * DK * SEQ;

  // Q^T B-fragments (reused across all key blocks)
  Frag16 qf[2];
#pragma unroll
  for (int st = 0; st < 2; ++st) {
    const unsigned short* qp = Q + (size_t)(q_base + l15) * DK + st * 32 + hi * 8;
    qf[st].q[0] = *(const uint4*)qp;
    qf[st].q[1] = *(const uint4*)(qp + 16);
  }

  v8f o[4];
#pragma unroll
  for (int n = 0; n < 4; ++n) o[n] = {};
  float m_run = -__builtin_inff();
  float l_run = 0.0f;

  const int nkb_w   = (q_base + 47) >> 5;                       // this wave
  const int nkb_blk = ((blockIdx.x * 4 + 3) * 16 + 47) >> 5;    // block max

  // cooperative stage of one 32-key block (4 async b128 per wave)
  auto stage = [&](int jb, int buf) {
#pragma unroll
    for (int r = 0; r < 2; ++r) {                    // K: 32 rows x 128B
      const int c = tid + r * 128;                   // 256 chunks of 16B
      const int row = c >> 3, ch = c & 7;
      g2l16(&Kb[buf][row * 72 + ch * 8], K + (size_t)(jb + row) * DK + ch * 8);
    }
#pragma unroll
    for (int r = 0; r < 2; ++r) {                    // V: 64 rows x 64B
      const int c = tid + r * 128;
      const int row = c >> 2, ch = c & 3;
      g2l16(&Vb[buf][row * 40 + ch * 8], V + (size_t)row * SEQ + jb + ch * 8);
    }
  };

  stage(0, 0);
  for (int i = 0; i < nkb_blk; ++i) {
    const int jb = i * 32;
    if (i + 1 < nkb_blk) { stage((i + 1) * 32, (i + 1) & 1); WAIT_ASYNC(4); }
    else                 { WAIT_ASYNC(0); }
    __syncthreads();                                 // block i data visible

    if (jb < nkb_w * 32) {                           // wave-uniform; EXEC full
      const unsigned short* Kl = Kb[i & 1];
      const unsigned short* Vl = Vb[i & 1];

      // ---- score tiles: S^T (M=key, N=query), 4 WMMAs ---------------------
      v8f sc[2]; sc[0] = {}; sc[1] = {};
#pragma unroll
      for (int t = 0; t < 2; ++t)
#pragma unroll
        for (int st = 0; st < 2; ++st) {
          Frag16 kf;
          const unsigned short* kp = Kl + (t * 16 + l15) * 72 + st * 32 + hi * 8;
          kf.q[0] = *(const uint4*)kp;
          kf.q[1] = *(const uint4*)(kp + 16);
          sc[t] = __builtin_amdgcn_wmma_f32_16x16x32_bf16(
              false, kf.v, false, qf[st].v, (short)0, sc[t], false, false);
        }

      // ---- causal mask (only near the diagonal) ---------------------------
      const int qg = q_base + l15;
      if (jb + 32 > q_base) {
#pragma unroll
        for (int t = 0; t < 2; ++t)
#pragma unroll
          for (int e = 0; e < 8; ++e) {
            const int kg = jb + t * 16 + e + hi * 8;
            if (kg > qg) sc[t][e] = -__builtin_inff();
          }
      }

      // ---- online softmax (per query = l15; halves joined via shfl 16) ----
      float vmax = -__builtin_inff();
#pragma unroll
      for (int t = 0; t < 2; ++t)
#pragma unroll
        for (int e = 0; e < 8; ++e) vmax = fmaxf(vmax, sc[t][e]);
      vmax = fmaxf(vmax, __shfl_xor(vmax, 16, 32));
      const float m_new = fmaxf(m_run, vmax);
      const float alpha = __expf(m_run - m_new);

      float rsum = 0.0f;
      Frag16 pa;                                 // P A-fragment, in-lane repack
#pragma unroll
      for (int t = 0; t < 2; ++t)
#pragma unroll
        for (int e = 0; e < 8; ++e) {
          const float p = __expf(sc[t][e] - m_new);
          rsum += p;
          pa.h[t * 8 + e] = f2bf(p);
        }
      rsum += __shfl_xor(rsum, 16, 32);
      l_run = l_run * alpha + rsum;
      m_run = m_new;

      // ---- rescale O rows by alpha (row index = e + hi*8 -> bpermute) -----
#pragma unroll
      for (int e = 0; e < 8; ++e) {
        const float ae = __shfl(alpha, e + hi * 8, 32);
#pragma unroll
        for (int n = 0; n < 4; ++n) o[n][e] *= ae;
      }

      // ---- O += P @ V  (V rows contiguous in LDS), 4 WMMAs ----------------
#pragma unroll
      for (int n = 0; n < 4; ++n) {
        Frag16 vf;
        const unsigned short* vp = Vl + (n * 16 + l15) * 40 + hi * 8;
        vf.q[0] = *(const uint4*)vp;
        vf.q[1] = *(const uint4*)(vp + 16);
        o[n] = __builtin_amdgcn_wmma_f32_16x16x32_bf16(
            false, pa.v, false, vf.v, (short)0, o[n], false, false);
      }
    }
    __syncthreads();                 // all waves done reading before overwrite
  }

  // ---- finalize: divide by l, store concat [b, s, h*64+d] bf16 ------------
#pragma unroll
  for (int e = 0; e < 8; ++e) {
    const float le  = __shfl(l_run, e + hi * 8, 32);
    const float inv = 1.0f / le;
    const int s = q_base + e + hi * 8;
#pragma unroll
    for (int n = 0; n < 4; ++n) {
      const int d = n * 16 + l15;
      concat[((size_t)b * SEQ + s) * DMODEL + h * DK + d] = f2bf(o[n][e] * inv);
    }
  }
}

// ---------------------------------------------------------------------------
// Kernel 3: output projection GEMM (bf16 WMMA) + bias -> f32 out.
// ---------------------------------------------------------------------------
__global__ __launch_bounds__(256)
void out_gemm(const unsigned short* __restrict__ A,    // [4096,1024] bf16
              const unsigned short* __restrict__ wT,   // [1024,1024] bf16 (N,K)
              const float* __restrict__ bias,
              float* __restrict__ out) {
  const int lane  = threadIdx.x & 31;
  const int wid   = threadIdx.x >> 5;
  const int waveM = wid >> 2, waveN = wid & 3;
  const int l15   = lane & 15;
  const int hi    = lane >> 4;
  const int rowA0 = blockIdx.y * 128 + waveM * 64;
  const int colB0 = blockIdx.x * 256 + waveN * 64;

  v8f acc[4][4];
#pragma unroll
  for (int mi = 0; mi < 4; ++mi)
#pragma unroll
    for (int ni = 0; ni < 4; ++ni) acc[mi][ni] = {};

  for (int kb = 0; kb < DMODEL; kb += 32) {
    Frag16 a[4];
#pragma unroll
    for (int mi = 0; mi < 4; ++mi) {
      const unsigned short* ap = A + (size_t)(rowA0 + mi * 16 + l15) * DMODEL
                                   + kb + hi * 8;
      a[mi].q[0] = *(const uint4*)ap;
      a[mi].q[1] = *(const uint4*)(ap + 16);
      if (kb + 32 < DMODEL) __builtin_prefetch(ap + 32, 0, 3);
    }
    Frag16 b[4];
#pragma unroll
    for (int ni = 0; ni < 4; ++ni) {
      const unsigned short* bp = wT + (size_t)(colB0 + ni * 16 + l15) * DMODEL
                                    + kb + hi * 8;
      b[ni].q[0] = *(const uint4*)bp;
      b[ni].q[1] = *(const uint4*)(bp + 16);
      if (kb + 32 < DMODEL) __builtin_prefetch(bp + 32, 0, 3);
    }
#pragma unroll
    for (int mi = 0; mi < 4; ++mi)
#pragma unroll
      for (int ni = 0; ni < 4; ++ni)
        acc[mi][ni] = __builtin_amdgcn_wmma_f32_16x16x32_bf16(
            false, a[mi].v, false, b[ni].v, (short)0, acc[mi][ni], false, false);
  }

#pragma unroll
  for (int mi = 0; mi < 4; ++mi)
#pragma unroll
    for (int ni = 0; ni < 4; ++ni) {
      const int n = colB0 + ni * 16 + l15;
      const float bv = bias[n];
#pragma unroll
      for (int e = 0; e < 8; ++e) {
        const int m = rowA0 + mi * 16 + e + hi * 8;
        out[(size_t)m * DMODEL + n] = acc[mi][ni][e] + bv;
      }
    }
}

// ---------------------------------------------------------------------------
extern "C" void kernel_launch(void* const* d_in, const int* in_sizes, int n_in,
                              void* d_out, int out_size, void* d_ws, size_t ws_size,
                              hipStream_t stream) {
  const float* x     = (const float*)d_in[0];
  const float* w_qkv = (const float*)d_in[1];
  const float* b_qkv = (const float*)d_in[2];
  const float* w_out = (const float*)d_in[3];
  const float* b_out = (const float*)d_in[4];
  float* out = (float*)d_out;

  char* ws = (char*)d_ws;
  size_t off = 0;
  unsigned short* wqkvT  = (unsigned short*)(ws + off); off += (size_t)NQKV * DMODEL * 2;
  unsigned short* woutT  = (unsigned short*)(ws + off); off += (size_t)DMODEL * DMODEL * 2;
  unsigned short* xb     = (unsigned short*)(ws + off); off += (size_t)MTOT * DMODEL * 2;
  unsigned short* qh     = (unsigned short*)(ws + off); off += (size_t)BATCH * HEADS * SEQ * DK * 2;
  unsigned short* kh     = (unsigned short*)(ws + off); off += (size_t)BATCH * HEADS * SEQ * DK * 2;
  unsigned short* vT     = (unsigned short*)(ws + off); off += (size_t)BATCH * HEADS * SEQ * DK * 2;
  unsigned short* concat = (unsigned short*)(ws + off); off += (size_t)MTOT * DMODEL * 2;
  (void)ws_size; (void)in_sizes; (void)n_in; (void)out_size;

  // 0) input cast + weight transposes (one-time, HBM-trivial)
  convert_bf16<<<(MTOT * DMODEL / 4 + 255) / 256, 256, 0, stream>>>(x, xb, MTOT * DMODEL / 4);
  transpose_to_bf16<<<(DMODEL * NQKV + 255) / 256, 256, 0, stream>>>(w_qkv, wqkvT, DMODEL, NQKV);
  transpose_to_bf16<<<(DMODEL * DMODEL + 255) / 256, 256, 0, stream>>>(w_out, woutT, DMODEL, DMODEL);

  // 1) QKV projection + bias + RoPE (Q scaled by sqrt(dk))
  qkv_gemm_rope<<<dim3(NQKV / 256, MTOT / 128), 256, 0, stream>>>(xb, wqkvT, b_qkv, qh, kh, vT);

  // 2) causal flash attention (async-LDS staged K/V)
  flash_attn<<<dim3(SEQ / 64, BATCH * HEADS), 128, 0, stream>>>(qh, kh, vT, concat);

  // 3) output projection
  out_gemm<<<dim3(DMODEL / 256, MTOT / 128), 256, 0, stream>>>(concat, woutT, b_out, out);
}
